// AudioCrossAttentionProcessor_28845000359990
// MI455X (gfx1250) — matrix operand order repaired and checked by reading, not compiled, not run on hardware
//
#include <hip/hip_runtime.h>

// ---------------------------------------------------------------------------
// AudioCrossAttentionProcessor for MI455X (gfx1250, wave32, WMMA bf16)
//
// Shapes: B=1 L1=10752 NF=21 L2=769 IMG=257 H=1536 N=12 D=128 L3=32 CA=768
// All matmuls on v_wmma_f32_16x16x32_bf16 (fp32 accum); RMS-norm/softmax in
// fp32 VALU. GEMM uses double-buffered ASYNCcnt-tracked global->LDS copies
// (global_load_async_to_lds_b128) when the toolchain exposes the builtins,
// else vectorized b128 load/store staging + global_prefetch_b8.
// ---------------------------------------------------------------------------

typedef __bf16 bf16;
typedef __bf16 v16bf __attribute__((ext_vector_type(16)));
typedef float  v8f   __attribute__((ext_vector_type(8)));

union FragB { v16bf v; bf16  e[16]; uint4 u[2]; };
union FragF { v8f   v; float e[8];  };

#if defined(__gfx1250__) && \
    __has_builtin(__builtin_amdgcn_global_load_async_to_lds_b128) && \
    __has_builtin(__builtin_amdgcn_s_wait_asynccnt)
#define USE_ASYNC_LDS 1
// builtin expects pointers to 16B int vectors: global (AS1) src, LDS (AS3) dst
typedef int v4i __attribute__((vector_size(16)));
typedef __attribute__((address_space(1))) v4i* as1_v4i;
typedef __attribute__((address_space(3))) v4i* as3_v4i;
#else
#define USE_ASYNC_LDS 0
#endif

// ISA 7.12.2 lane layouts (lane = lo + 16*hi):
//  A (16x32, 16-bit): half h -> K = ((h&8)<<1) + hi*8 + (h&7)  => two 8-half
//    contiguous runs at K = hi*8 and K = 16+hi*8.
//  B (32x16, 16-bit): same K order as A. This code instead uses K = (h&15) + hi*16 for A and B
//    (one 16-half contiguous run at K = hi*16); valid since A and B match.
//  C/D (16x16 f32):  elem j -> (M = j + 8*hi, N = lo).

// ---------------------------------------------------------------------------
// Elementwise fp32 -> bf16
// ---------------------------------------------------------------------------
__global__ void cvt_f32_bf16(const float* __restrict__ in, bf16* __restrict__ out, long long n) {
    long long i = (long long)blockIdx.x * blockDim.x + threadIdx.x;
    if (i < n) out[i] = (bf16)in[i];
}

__global__ void fill_zero_bf16(bf16* __restrict__ p, long long n) {
    long long i = (long long)blockIdx.x * blockDim.x + threadIdx.x;
    if (i < n) p[i] = (bf16)0.0f;
}

// ---------------------------------------------------------------------------
// Per-row RMS-norm (optional) + per-column weight (optional) -> bf16.
// transp=0: out[row*ncol+i]; transp=1: out[i*ostride + row] (for V^T).
// ---------------------------------------------------------------------------
__global__ void rms_cvt(const float* __restrict__ in, const float* __restrict__ w,
                        bf16* __restrict__ out, int ncol, int do_norm,
                        int transp, int ostride) {
    int row = blockIdx.x;
    const float* xr = in + (long long)row * ncol;
    __shared__ float red[256];
    float s = 0.f;
    for (int i = threadIdx.x; i < ncol; i += 256) { float v = xr[i]; s += v * v; }
    red[threadIdx.x] = s;
    __syncthreads();
    for (int st = 128; st > 0; st >>= 1) {
        if (threadIdx.x < st) red[threadIdx.x] += red[threadIdx.x + st];
        __syncthreads();
    }
    float scale = do_norm ? rsqrtf(red[0] / (float)ncol + 1e-6f) : 1.0f;
    for (int i = threadIdx.x; i < ncol; i += 256) {
        float v = xr[i] * scale;
        if (w) v *= w[i];
        if (transp) out[(long long)i * ostride + row] = (bf16)v;
        else        out[(long long)row * ncol + i]    = (bf16)v;
    }
}

// ---------------------------------------------------------------------------
// C[M,Nc] = A[M,K] * W[Nc,K]^T (+bias), bf16 operands, fp32 output.
// Block: 256 thr = 8 waves; block tile 128x128, wave tile 32x64 (2x4 WMMA).
// Double-buffered LDS; async global->LDS when available.
// ---------------------------------------------------------------------------
#define BM 128
#define BN 128
#define BK 32
#define LDK (BK + 8)   // 40 elems = 80B rows: 16B-aligned

__launch_bounds__(256)
__global__ void gemm_bf16_wmma(const bf16* __restrict__ A, const bf16* __restrict__ W,
                               const float* __restrict__ bias, float* __restrict__ C,
                               int M, int Nc, int K) {
    __shared__ __align__(16) bf16 As[2][BM][LDK];
    __shared__ __align__(16) bf16 Ws[2][BM][LDK];
    const int tid  = threadIdx.x;
    const int wave = tid >> 5;
    const int lane = tid & 31;
    const int lo   = lane & 15, hi = lane >> 4;
    const int wm   = wave >> 1;   // 0..3 -> m offset 32*wm
    const int wn   = wave & 1;    // 0..1 -> n offset 64*wn
    const int blockM = blockIdx.x * BM;
    const int blockN = blockIdx.y * BN;

    // this thread's two staging slots (r, c8) covering 512 uint4 per operand
    const int r0  = tid >> 2;            // 0..63
    const int r1  = r0 + 64;             // 64..127
    const int c8  = (tid & 3) * 8;       // 0,8,16,24
    // clamp M-tail rows to a valid address; garbage only feeds unstored C rows
    const int ga0 = (blockM + r0 < M) ? (blockM + r0) : (M - 1);
    const int ga1 = (blockM + r1 < M) ? (blockM + r1) : (M - 1);

    FragF acc[2][4] = {};
    const int nt = K / BK;

#if USE_ASYNC_LDS
    // issue one K-tile's staging: 4 async b128 copies per thread/wave
    auto issue_tile = [&](int k0, int buf) {
        __builtin_amdgcn_global_load_async_to_lds_b128(
            (as1_v4i)(A + (long long)ga0 * K + k0 + c8),
            (as3_v4i)&As[buf][r0][c8], 0, 0);
        __builtin_amdgcn_global_load_async_to_lds_b128(
            (as1_v4i)(A + (long long)ga1 * K + k0 + c8),
            (as3_v4i)&As[buf][r1][c8], 0, 0);
        __builtin_amdgcn_global_load_async_to_lds_b128(
            (as1_v4i)(W + (long long)(blockN + r0) * K + k0 + c8),
            (as3_v4i)&Ws[buf][r0][c8], 0, 0);
        __builtin_amdgcn_global_load_async_to_lds_b128(
            (as1_v4i)(W + (long long)(blockN + r1) * K + k0 + c8),
            (as3_v4i)&Ws[buf][r1][c8], 0, 0);
    };
    issue_tile(0, 0);
#endif

    for (int i = 0; i < nt; i++) {
        const int buf = i & 1;
        const int k0  = i * BK;
#if USE_ASYNC_LDS
        if (i + 1 < nt) {
            issue_tile(k0 + BK, buf ^ 1);
            __builtin_amdgcn_s_wait_asynccnt(4);  // tile i complete (in-order retire)
        } else {
            __builtin_amdgcn_s_wait_asynccnt(0);
        }
        __syncthreads();
#else
        *(uint4*)&As[buf][r0][c8] = *(const uint4*)(A + (long long)ga0 * K + k0 + c8);
        *(uint4*)&As[buf][r1][c8] = *(const uint4*)(A + (long long)ga1 * K + k0 + c8);
        *(uint4*)&Ws[buf][r0][c8] = *(const uint4*)(W + (long long)(blockN + r0) * K + k0 + c8);
        *(uint4*)&Ws[buf][r1][c8] = *(const uint4*)(W + (long long)(blockN + r1) * K + k0 + c8);
        if (k0 + BK < K) {
            __builtin_prefetch(A + (long long)ga0 * K + k0 + BK + c8, 0, 1);
            __builtin_prefetch(W + (long long)(blockN + r0) * K + k0 + BK + c8, 0, 1);
        }
        __syncthreads();
#endif

        FragB afr[2], bfr[4];
        for (int ms = 0; ms < 2; ms++) {
            int row = wm * 32 + ms * 16 + lo;
            afr[ms].u[0] = *(const uint4*)&As[buf][row][hi * 16];
            afr[ms].u[1] = *(const uint4*)&As[buf][row][hi * 16 + 8];
        }
        for (int ns = 0; ns < 4; ns++) {
            int col = wn * 64 + ns * 16 + lo;
            bfr[ns].u[0] = *(const uint4*)&Ws[buf][col][hi * 16];
            bfr[ns].u[1] = *(const uint4*)&Ws[buf][col][hi * 16 + 8];
        }
        for (int ms = 0; ms < 2; ms++)
            for (int ns = 0; ns < 4; ns++)
                acc[ms][ns].v = __builtin_amdgcn_wmma_f32_16x16x32_bf16(
                    false, afr[ms].v, false, bfr[ns].v, (short)0, acc[ms][ns].v, false, false);
        __syncthreads();
    }

    for (int ms = 0; ms < 2; ms++) {
        for (int ns = 0; ns < 4; ns++) {
            int n   = blockN + wn * 64 + ns * 16 + lo;
            float b = bias ? bias[n] : 0.0f;
            for (int j = 0; j < 8; j++) {
                int m = blockM + wm * 32 + ms * 16 + j + hi * 8;
                if (m < M) C[(long long)m * Nc + n] = acc[ms][ns].e[j] + b;
            }
        }
    }
}

// ---------------------------------------------------------------------------
// Flash attention. q,k laid out [rows, 1536]; V is TRANSPOSED: vT[d][key]
// with key-stride `kstride` (padded to a multiple of 32, zero-filled).
// One wave per (16-query tile, head). 32-key tiles: 8 WMMA score + 8 WMMA PV.
// mode: 0 = img (no mask, store), 1 = txt (klens[0] mask, accumulate),
//       2 = audio (per-frame 32-key segment, klens[frame] mask, accumulate)
// ---------------------------------------------------------------------------
__launch_bounds__(32)
__global__ void flash_attn_bf16(const bf16* __restrict__ q, const bf16* __restrict__ kmat,
                                const bf16* __restrict__ vT, float* __restrict__ y,
                                int Lk, int kstride, const int* __restrict__ klens, int mode) {
    const int Hdim = 1536;
    const int qt = blockIdx.x, head = blockIdx.y;
    const int lane = threadIdx.x, lo = lane & 15, hi = lane >> 4;

    int kbase = 0, kvalid;
    if (mode == 2) { int fr = qt >> 5; kbase = fr * Lk; kvalid = min(Lk, klens[fr]); }
    else if (mode == 1) kvalid = min(Lk, klens[0]);
    else kvalid = Lk;

    const long long qrow0 = (long long)qt * 16;
    const long long hoff  = (long long)head * 128;

    // q fragments: 4 K-chunks of 32, two contiguous 16B runs each
    FragB aq[4];
    {
        const bf16* qb = q + (qrow0 + lo) * Hdim + hoff;
        for (int c = 0; c < 4; c++) {
            aq[c].u[0] = *(const uint4*)(qb + c * 32 + hi * 16);
            aq[c].u[1] = *(const uint4*)(qb + c * 32 + hi * 16 + 8);
        }
    }

    float mrow[8], lrow[8];
    FragF Ov[8];
    for (int j = 0; j < 8; j++) { mrow[j] = -3.0e38f; lrow[j] = 0.f; }
    for (int d = 0; d < 8; d++) Ov[d] = FragF{};

    __shared__ __align__(16) bf16 Plds[16][32 + 8];
    const float sscale = 0.08838834764831845f; // 1/sqrt(128)

    const int ntiles = (Lk + 31) / 32;
    for (int t = 0; t < ntiles; t++) {
        // ---- scores S = q K^T for 32 keys (2 halves of 16) ----
        FragF S[2];
        for (int half = 0; half < 2; half++) {
            FragF s = {};
            int key = t * 32 + half * 16 + lo;  // this lane's key column
            for (int c = 0; c < 4; c++) {
                FragB bk = {};
                if (key < Lk) {
                    const bf16* kb = kmat + (long long)(kbase + key) * Hdim + hoff
                                   + c * 32 + hi * 16;
                    bk.u[0] = *(const uint4*)kb;
                    bk.u[1] = *(const uint4*)(kb + 8);
                }
                s.v = __builtin_amdgcn_wmma_f32_16x16x32_bf16(
                    false, aq[c].v, false, bk.v, (short)0, s.v, false, false);
            }
            if (key < kvalid) { for (int j = 0; j < 8; j++) s.e[j] *= sscale; }
            else              { for (int j = 0; j < 8; j++) s.e[j] = -1e30f; }
            S[half] = s;
        }

        // ---- online softmax (row reductions across 16-lane groups) ----
        float rmax[8];
        for (int j = 0; j < 8; j++) rmax[j] = fmaxf(S[0].e[j], S[1].e[j]);
        for (int msk = 1; msk < 16; msk <<= 1)
            for (int j = 0; j < 8; j++) rmax[j] = fmaxf(rmax[j], __shfl_xor(rmax[j], msk, 32));

        float corr[8];
        for (int j = 0; j < 8; j++) {
            float mn = fmaxf(mrow[j], rmax[j]);
            corr[j]  = __expf(mrow[j] - mn);
            mrow[j]  = mn;
        }
        float rsum[8];
        for (int j = 0; j < 8; j++) {
            float p0 = __expf(S[0].e[j] - mrow[j]);
            float p1 = __expf(S[1].e[j] - mrow[j]);
            S[0].e[j] = p0; S[1].e[j] = p1;
            rsum[j] = p0 + p1;
        }
        for (int msk = 1; msk < 16; msk <<= 1)
            for (int j = 0; j < 8; j++) rsum[j] += __shfl_xor(rsum[j], msk, 32);
        for (int j = 0; j < 8; j++) lrow[j] = lrow[j] * corr[j] + rsum[j];
        for (int d = 0; d < 8; d++)
            for (int j = 0; j < 8; j++) Ov[d].e[j] *= corr[j];

        // ---- restage P (C-layout fp32) through LDS into A-layout bf16 ----
        __syncthreads();
        for (int half = 0; half < 2; half++)
            for (int j = 0; j < 8; j++)
                Plds[j + hi * 8][half * 16 + lo] = (bf16)S[half].e[j];
        __syncthreads();
        FragB Pa;
        Pa.u[0] = *(const uint4*)&Plds[lo][hi * 16];
        Pa.u[1] = *(const uint4*)&Plds[lo][hi * 16 + 8];

        // ---- O += P V : B operand from transposed V, contiguous per lane ----
        for (int d = 0; d < 8; d++) {
            FragB bv;
            const bf16* vb = vT + (long long)(hoff + d * 16 + lo) * kstride
                           + kbase + t * 32 + hi * 16;
            bv.u[0] = *(const uint4*)vb;
            bv.u[1] = *(const uint4*)(vb + 8);
            Ov[d].v = __builtin_amdgcn_wmma_f32_16x16x32_bf16(
                false, Pa.v, false, bv.v, (short)0, Ov[d].v, false, false);
        }
        __syncthreads();
    }

    for (int j = 0; j < 8; j++) {
        float inv = 1.0f / lrow[j];
        long long row = qrow0 + j + hi * 8;
        for (int d = 0; d < 8; d++) {
            long long idx = row * Hdim + hoff + d * 16 + lo;
            float val = Ov[d].e[j] * inv;
            if (mode == 0) y[idx] = val; else y[idx] += val;
        }
    }
}

// ---------------------------------------------------------------------------
// Host orchestration
// ---------------------------------------------------------------------------
extern "C" void kernel_launch(void* const* d_in, const int* in_sizes, int n_in,
                              void* d_out, int out_size, void* d_ws, size_t ws_size,
                              hipStream_t stream) {
    const int L1 = 10752, L2 = 769, IMG = 257, H = 1536, NF = 21, L3 = 32, CA = 768;
    const int TXT = L2 - IMG;        // 512
    const int AUD = NF * L3;         // 672
    const int IMGP = 288;            // img keys padded to multiple of 32

    const float* x    = (const float*)d_in[0];
    const float* ctx  = (const float*)d_in[1];
    const float* aud  = (const float*)d_in[2];
    const float* Wq   = (const float*)d_in[3];
    const float* bq   = (const float*)d_in[4];
    const float* Wk   = (const float*)d_in[5];
    const float* bk   = (const float*)d_in[6];
    const float* Wv   = (const float*)d_in[7];
    const float* bv   = (const float*)d_in[8];
    const float* Wki  = (const float*)d_in[9];
    const float* bki  = (const float*)d_in[10];
    const float* Wvi  = (const float*)d_in[11];
    const float* bvi  = (const float*)d_in[12];
    const float* Wo   = (const float*)d_in[13];
    const float* bo   = (const float*)d_in[14];
    const float* nqw  = (const float*)d_in[15];
    const float* nkw  = (const float*)d_in[16];
    const float* nkiw = (const float*)d_in[17];
    const float* Wkp  = (const float*)d_in[18];
    const float* Wvp  = (const float*)d_in[19];
    const int*   clen = (const int*)d_in[20];
    const int*   alen = (const int*)d_in[21];
    float* out = (float*)d_out;

    char* wsb = (char*)d_ws;
    size_t off = 0;
    auto alloc = [&](size_t bytes) -> char* {
        char* p = wsb + off;
        off = (off + bytes + 255) & ~(size_t)255;
        return p;
    };

    bf16* xb    = (bf16*)alloc((size_t)L1 * H * 2);    // x bf16; reused as y_bf16 later
    bf16* ctxb  = (bf16*)alloc((size_t)L2 * H * 2);
    bf16* audb  = (bf16*)alloc((size_t)AUD * CA * 2);
    bf16* wqb   = (bf16*)alloc((size_t)H * H * 2);
    bf16* wkb   = (bf16*)alloc((size_t)H * H * 2);
    bf16* wvb   = (bf16*)alloc((size_t)H * H * 2);
    bf16* wkib  = (bf16*)alloc((size_t)H * H * 2);
    bf16* wvib  = (bf16*)alloc((size_t)H * H * 2);
    bf16* wob   = (bf16*)alloc((size_t)H * H * 2);
    bf16* wkpb  = (bf16*)alloc((size_t)H * CA * 2);
    bf16* wvpb  = (bf16*)alloc((size_t)H * CA * 2);
    float* lin  = (float*)alloc((size_t)L1 * H * 4);   // fp32 projection scratch (reused)
    bf16* qbf   = (bf16*)alloc((size_t)L1 * H * 2);
    bf16* kbf   = (bf16*)alloc((size_t)TXT * H * 2);
    bf16* vbfT  = (bf16*)alloc((size_t)H * TXT * 2);   // V^T [1536][512]
    bf16* kib   = (bf16*)alloc((size_t)IMG * H * 2);
    bf16* vibT  = (bf16*)alloc((size_t)H * IMGP * 2);  // V_img^T [1536][288] zero-padded
    bf16* ipk   = (bf16*)alloc((size_t)AUD * H * 2);
    bf16* ipvT  = (bf16*)alloc((size_t)H * AUD * 2);   // V_aud^T [1536][672]
    float* yac  = (float*)alloc((size_t)L1 * H * 4);

    auto cvt = [&](const float* src, bf16* dst, long long n) {
        cvt_f32_bf16<<<(int)((n + 255) / 256), 256, 0, stream>>>(src, dst, n);
    };
    auto gemm = [&](const bf16* A, const bf16* W, const float* bias, float* C,
                    int M, int Nc, int K) {
        dim3 g((M + BM - 1) / BM, Nc / BN);
        gemm_bf16_wmma<<<g, 256, 0, stream>>>(A, W, bias, C, M, Nc, K);
    };
    auto rms = [&](const float* in, const float* w, bf16* o, int rows, int ncol,
                   int norm, int transp, int ostride) {
        rms_cvt<<<rows, 256, 0, stream>>>(in, w, o, ncol, norm, transp, ostride);
    };

    // 0) down-convert activations + weights to bf16
    cvt(x, xb, (long long)L1 * H);
    cvt(ctx, ctxb, (long long)L2 * H);
    cvt(aud, audb, (long long)AUD * CA);
    cvt(Wq, wqb, (long long)H * H);
    cvt(Wk, wkb, (long long)H * H);
    cvt(Wv, wvb, (long long)H * H);
    cvt(Wki, wkib, (long long)H * H);
    cvt(Wvi, wvib, (long long)H * H);
    cvt(Wo, wob, (long long)H * H);
    cvt(Wkp, wkpb, (long long)H * CA);
    cvt(Wvp, wvpb, (long long)H * CA);
    fill_zero_bf16<<<(int)(((long long)H * IMGP + 255) / 256), 256, 0, stream>>>(
        vibT, (long long)H * IMGP);

    // 1) projections (+RMS-norm where needed); V projections stored transposed
    gemm(xb, wqb, bq, lin, L1, H, H);                       // q = x Wq^T + bq
    rms(lin, nqw, qbf, L1, H, 1, 0, 0);
    gemm(ctxb + (size_t)IMG * H, wkb, bk, lin, TXT, H, H);  // k (text)
    rms(lin, nkw, kbf, TXT, H, 1, 0, 0);
    gemm(ctxb + (size_t)IMG * H, wvb, bv, lin, TXT, H, H);  // v (text) -> V^T
    rms(lin, nullptr, vbfT, TXT, H, 0, 1, TXT);
    gemm(ctxb, wkib, bki, lin, IMG, H, H);                  // k_img
    rms(lin, nkiw, kib, IMG, H, 1, 0, 0);
    gemm(ctxb, wvib, bvi, lin, IMG, H, H);                  // v_img -> V^T (padded)
    rms(lin, nullptr, vibT, IMG, H, 0, 1, IMGP);
    gemm(audb, wkpb, nullptr, lin, AUD, H, CA);             // ip_key
    rms(lin, nullptr, ipk, AUD, H, 0, 0, 0);
    gemm(audb, wvpb, nullptr, lin, AUD, H, CA);             // ip_value -> V^T
    rms(lin, nullptr, ipvT, AUD, H, 0, 1, AUD);

    // 2) three attentions accumulating into yac
    dim3 ag(L1 / 16, 12);
    flash_attn_bf16<<<ag, 32, 0, stream>>>(qbf, kib, vibT, yac, IMG, IMGP, nullptr, 0);
    flash_attn_bf16<<<ag, 32, 0, stream>>>(qbf, kbf, vbfT, yac, TXT, TXT, clen, 1);
    flash_attn_bf16<<<ag, 32, 0, stream>>>(qbf, ipk, ipvT, yac, L3, AUD, alen, 2);

    // 3) y -> bf16 (reuse xb), final out-proj straight to d_out (fp32 + bias)
    rms(yac, nullptr, xb, L1, H, 0, 0, 0);
    gemm(xb, wob, bo, out, L1, H, H);
}
